// AxisNet_35519379538313
// MI455X (gfx1250) — matrix operand order repaired and checked by reading, not compiled, not run on hardware
//
#include <hip/hip_runtime.h>
#include <hip/hip_bf16.h>

// ---------------------------------------------------------------------------
// AxisNet for MI455X (gfx1250, wave32).
// One wave per point, 8 waves / 256-thread block. Per-point 32x9 -> 16 -> 32
// -> 64 MLP mapped onto v_wmma_f32_16x16x32_f16 (14 WMMAs per wave), K-maxpool
// on D fragments, small head MLP on VALU. Weights cached in LDS; fW1 is
// brought in via the Tensor Data Mover (tensor_load_to_lds + s_wait_tensorcnt).
// ---------------------------------------------------------------------------

typedef __attribute__((ext_vector_type(16))) _Float16 v16h;
typedef __attribute__((ext_vector_type(8)))  _Float16 v8h;
typedef __attribute__((ext_vector_type(8)))  float    v8f;
typedef __attribute__((ext_vector_type(4)))  unsigned int v4u;
typedef __attribute__((ext_vector_type(8)))  int      v8i;
typedef __attribute__((ext_vector_type(4)))  int      v4i;

#define BATCH 4
#define NPTS  8192
#define KNBR  32
#define BN    (BATCH * NPTS)
#define WVB   8          // waves per block
#define SST   40         // stage row stride in halfs (80B, 16B aligned)

#if defined(__has_builtin)
#if __has_builtin(__builtin_amdgcn_tensor_load_to_lds) && __has_builtin(__builtin_amdgcn_s_wait_tensorcnt)
#define HAVE_TDM 1
#endif
#endif

__device__ __forceinline__ v8f wmma_f16(v16h a, v16h b, v8f c) {
  // (neg_a, A, neg_b, B, c_mod, C, reuse_a, reuse_b)
  return __builtin_amdgcn_wmma_f32_16x16x32_f16(false, a, false, b, (short)0, c,
                                                false, false);
}

__global__ __launch_bounds__(256) void axisnet_kernel(
    const float* __restrict__ xyz, const float* __restrict__ local_axis,
    const float* __restrict__ W1, const float* __restrict__ b1,
    const float* __restrict__ W2, const float* __restrict__ b2,
    const float* __restrict__ W3, const float* __restrict__ b3,
    const float* __restrict__ fW1, const float* __restrict__ fb1,
    const float* __restrict__ fW2, const float* __restrict__ fb2,
    const float* __restrict__ fW3, const float* __restrict__ fb3,
    const int* __restrict__ nbr, float* __restrict__ out)
{
  // ---- LDS: weights (block-shared) + per-wave staging tiles ----
  __shared__ __attribute__((aligned(64))) _Float16 w1t[16][16];   // [n][k], k>=9 zero
  __shared__ __attribute__((aligned(64))) _Float16 w2t[32][16];   // [n][k]
  __shared__ __attribute__((aligned(64))) _Float16 w3t[64][32];   // [n][k]
  __shared__ __attribute__((aligned(64))) float b1f[16];
  __shared__ __attribute__((aligned(64))) float b2f[32];
  __shared__ __attribute__((aligned(64))) float b3f[64];
  __shared__ __attribute__((aligned(64))) float fw1f[64 * 32];    // TDM target
  __shared__ __attribute__((aligned(64))) float fw2f[32 * 32];
  __shared__ __attribute__((aligned(64))) float fb1f[32];
  __shared__ __attribute__((aligned(64))) float fb2f[32];
  __shared__ __attribute__((aligned(64))) float fw3f[32];
  __shared__ __attribute__((aligned(64))) float fb3f[4];
  __shared__ __attribute__((aligned(64))) _Float16 stage[WVB][32][SST];
  __shared__ __attribute__((aligned(64))) float l3buf[WVB][64];
  __shared__ __attribute__((aligned(64))) float y1buf[WVB][32];

  const int tid  = threadIdx.x;
  const int lane = tid & 31;
  const int wv   = tid >> 5;
  const int p    = blockIdx.x * WVB + wv;          // point id, grid sized exactly
  const int bb   = p >> 13;                        // batch index (N = 8192)
  const bool lo  = lane < 16;
  const int  nl  = lane & 15;

  // ---- gather neighbor data early (one neighbor per lane) ----
  const int   idx = nbr[(size_t)p * KNBR + lane];
  const float* gp = xyz + (size_t)(bb * NPTS + idx) * 6;
  const float gx = gp[0], gy = gp[1], gz = gp[2];
  const float hx = gp[3], hy = gp[4], hz = gp[5];
  const float* cp = xyz + (size_t)p * 6;
  const float cx = cp[0], cy = cp[1], cz = cp[2];
  const float* Ax = local_axis + (size_t)p * 9;    // A[c][d] = Ax[c*3+d]
  const float a00 = Ax[0], a01 = Ax[1], a02 = Ax[2];
  const float a10 = Ax[3], a11 = Ax[4], a12 = Ax[5];
  const float a20 = Ax[6], a21 = Ax[7], a22 = Ax[8];

  // ---- cooperative weight staging into LDS ----
  {
    int n = tid >> 4, k = tid & 15;
    w1t[n][k] = (k < 9) ? (_Float16)W1[k * 16 + n] : (_Float16)0.0f;
  }
  for (int i = tid; i < 512; i += 256) {
    int n = i >> 4, k = i & 15;
    w2t[n][k] = (_Float16)W2[k * 32 + n];
  }
  for (int i = tid; i < 2048; i += 256) {
    int n = i >> 5, k = i & 31;
    w3t[n][k] = (_Float16)W3[k * 64 + n];
  }
  if (tid < 16) b1f[tid] = b1[tid];
  if (tid < 32) {
    b2f[tid]  = b2[tid];
    fb1f[tid] = fb1[tid];
    fb2f[tid] = fb2[tid];
    fw3f[tid] = fW3[tid];
  }
  if (tid < 64) b3f[tid] = b3[tid];
  if (tid == 0) fb3f[0] = fb3[0];
  for (int i = tid; i < 1024; i += 256) fw2f[i] = fW2[i];

#if HAVE_TDM
  // Tensor Data Mover: copy fW1 (64x32 f32 = 8KB, contiguous) into LDS.
  // D# per ISA ch.8: group0 = {count=1, lds_addr, global_addr, type=2};
  // group1 = {data_size=4B, tensor_dim0=2048, tensor_dim1=1, tile=2048x1,
  //           dim0_stride=2048}. Issued by wave 0 only (TENSORcnt is per-wave).
  if (tid < 32) {
    unsigned long long ga = (unsigned long long)fW1;
    unsigned lds_off = (unsigned)(unsigned long long)&fw1f[0]; // addr[31:0] = LDS offset
    v4u g0;
    g0[0] = 1u;                                            // count=1
    g0[1] = lds_off;                                       // lds_addr
    g0[2] = (unsigned)(ga & 0xFFFFFFFFu);                  // global_addr lo
    g0[3] = (unsigned)((ga >> 32) & 0x01FFFFFFu) | (2u << 30); // addr hi | type=2
    v8i g1;
    g1[0] = (int)(2u << 16);        // data_size = 2 (4 bytes)
    g1[1] = (int)(2048u << 16);     // tensor_dim0 = 2048 (low 16 in [31:16])
    g1[2] = (int)(1u << 16);        // tensor_dim1 = 1
    g1[3] = (int)(2048u << 16);     // tile_dim0 = 2048
    g1[4] = 1;                      // tile_dim1 = 1, tile_dim2 = 0
    g1[5] = 2048;                   // tensor_dim0_stride
    g1[6] = 0;
    g1[7] = 0;
    v4i gz4 = {0, 0, 0, 0};
#if __clang_major__ >= 23
    v8i gz8 = {0, 0, 0, 0, 0, 0, 0, 0};
    __builtin_amdgcn_tensor_load_to_lds(g0, g1, gz4, gz4, gz8, 0);
#else
    __builtin_amdgcn_tensor_load_to_lds(g0, g1, gz4, gz4, 0);
#endif
    __builtin_amdgcn_s_wait_tensorcnt(0);
  }
#else
  for (int i = tid; i < 2048; i += 256) fw1f[i] = fW1[i];
#endif
  __syncthreads();

  // ---- per-lane features: [local_xyz(3) | grouped_xyz(3) | rotated_nrm(3)] ----
  const float lx = gx - cx, ly = gy - cy, lz = gz - cz;
  const float f0 = lx * a00 + ly * a10 + lz * a20;
  const float f1 = lx * a01 + ly * a11 + lz * a21;
  const float f2 = lx * a02 + ly * a12 + lz * a22;
  const float f6 = hx * a00 + hy * a10 + hz * a20;
  const float f7 = hx * a01 + hy * a11 + hz * a21;
  const float f8 = hx * a02 + hy * a12 + hz * a22;

  {
    v8h z8 = {};
    _Float16* row = &stage[wv][lane][0];
    *(v8h*)(row + 0)  = z8;          // zero K-padding (cols 0..31)
    *(v8h*)(row + 8)  = z8;
    *(v8h*)(row + 16) = z8;
    *(v8h*)(row + 24) = z8;
    row[0] = (_Float16)f0; row[1] = (_Float16)f1; row[2] = (_Float16)f2;
    row[3] = (_Float16)gx; row[4] = (_Float16)gy; row[5] = (_Float16)gz;
    row[6] = (_Float16)f6; row[7] = (_Float16)f7; row[8] = (_Float16)f8;
  }

  // A fragment (16x32 f16): lanes 0-15 take K 0-15, lanes 16-31 take
  // K 16-31 -> two contiguous 16B LDS chunks per lane.
  auto build_a = [&](int m0) -> v16h {
    const _Float16* r = &stage[wv][m0 + nl][0];
    const int kb = lo ? 0 : 16;
    v8h c0 = *(const v8h*)(r + kb);
    v8h c1 = *(const v8h*)(r + kb + 8);
    return __builtin_shufflevector(c0, c1, 0, 1, 2, 3, 4, 5, 6, 7,
                                   8, 9, 10, 11, 12, 13, 14, 15);
  };

  const v8f zacc = {};
  const v16h zb  = {};

  // ================= layer 1: (32x32pad) x (32x16) =================
  v16h aL = build_a(0), aH = build_a(16);
  v16h bw1 = lo ? *(const v16h*)&w1t[nl][0] : zb;   // K 16..31 are zero rows
  v8f d0 = wmma_f16(aL, bw1, zacc);
  v8f d1 = wmma_f16(aH, bw1, zacc);
  {
    const float bias = b1f[nl];
#pragma unroll
    for (int v = 0; v < 8; ++v) {
      const int r = lo ? v : v + 8;
      stage[wv][r][nl]      = (_Float16)fmaxf(d0[v] + bias, 0.0f);
      stage[wv][16 + r][nl] = (_Float16)fmaxf(d1[v] + bias, 0.0f);
    }
  }

  // ================= layer 2: (32x32pad) x (32x32) =================
  aL = build_a(0); aH = build_a(16);
  v16h b20 = lo ? *(const v16h*)&w2t[nl][0]      : zb;
  v16h b21 = lo ? *(const v16h*)&w2t[16 + nl][0] : zb;
  v8f e00 = wmma_f16(aL, b20, zacc);
  v8f e10 = wmma_f16(aH, b20, zacc);
  v8f e01 = wmma_f16(aL, b21, zacc);
  v8f e11 = wmma_f16(aH, b21, zacc);
#pragma unroll
  for (int nt = 0; nt < 2; ++nt) {
    const float bias = b2f[nt * 16 + nl];
    v8f eL = nt ? e01 : e00;
    v8f eH = nt ? e11 : e10;
#pragma unroll
    for (int v = 0; v < 8; ++v) {
      const int r = lo ? v : v + 8;
      stage[wv][r][nt * 16 + nl]      = (_Float16)fmaxf(eL[v] + bias, 0.0f);
      stage[wv][16 + r][nt * 16 + nl] = (_Float16)fmaxf(eH[v] + bias, 0.0f);
    }
  }

  // ============ layer 3: (32x32) x (32x64) + maxpool over K ============
  aL = build_a(0); aH = build_a(16);
#pragma unroll
  for (int nt = 0; nt < 4; ++nt) {
    v16h b3w = *(const v16h*)(&w3t[nt * 16 + nl][0] + (lo ? 0 : 16));
    v8f q0 = wmma_f16(aL, b3w, zacc);
    v8f q1 = wmma_f16(aH, b3w, zacc);
    float mx = fmaxf(q0[0], q1[0]);
#pragma unroll
    for (int v = 1; v < 8; ++v) mx = fmaxf(mx, fmaxf(q0[v], q1[v]));
    mx = fmaxf(mx, __shfl_xor(mx, 16, 32));            // M 0-7 vs 8-15 halves
    const float l3v = fmaxf(mx + b3f[nt * 16 + nl], 0.0f);
    if (lo) l3buf[wv][nt * 16 + nl] = l3v;
  }

  // ================= head MLP: 64 -> 32 -> 32 -> 1 =================
  float acc = fb1f[lane];
#pragma unroll
  for (int i = 0; i < 64; ++i) acc += l3buf[wv][i] * fw1f[i * 32 + lane];
  acc = fmaxf(acc, 0.0f);
  y1buf[wv][lane] = acc;

  float acc2 = fb2f[lane];
#pragma unroll
  for (int i = 0; i < 32; ++i) acc2 += y1buf[wv][i] * fw2f[i * 32 + lane];
  acc2 = fmaxf(acc2, 0.0f);

  float part = acc2 * fw3f[lane];
#pragma unroll
  for (int m = 16; m >= 1; m >>= 1) part += __shfl_xor(part, m, 32);

  const float ang = tanhf(part + fb3f[0]) * 6.283185306f;  // 2 * 3.141592653
  const float cv = cosf(ang), sv = sinf(ang);

  if (lane == 0) {
    const float nx = cp[3], ny = cp[4], nz = cp[5];
    const float xa0 = cv * a00 + sv * a01;
    const float xa1 = cv * a10 + sv * a11;
    const float xa2 = cv * a20 + sv * a21;
    const float ya0 = ny * xa2 - nz * xa1;   // y = n x x_axis
    const float ya1 = nz * xa0 - nx * xa2;
    const float ya2 = nx * xa1 - ny * xa0;
    float* ox = out + (size_t)p * 3;
    float* oy = out + (size_t)BN * 3 + (size_t)p * 3;
    float* on = out + (size_t)BN * 6 + (size_t)p * 3;
    ox[0] = xa0; ox[1] = xa1; ox[2] = xa2;
    oy[0] = ya0; oy[1] = ya1; oy[2] = ya2;
    on[0] = nx;  on[1] = ny;  on[2] = nz;
  }
}

extern "C" void kernel_launch(void* const* d_in, const int* in_sizes, int n_in,
                              void* d_out, int out_size, void* d_ws, size_t ws_size,
                              hipStream_t stream) {
  (void)in_sizes; (void)n_in; (void)out_size; (void)d_ws; (void)ws_size;
  const float* xyz        = (const float*)d_in[0];
  const float* local_axis = (const float*)d_in[1];
  const float* W1  = (const float*)d_in[2];
  const float* b1  = (const float*)d_in[3];
  const float* W2  = (const float*)d_in[4];
  const float* b2  = (const float*)d_in[5];
  const float* W3  = (const float*)d_in[6];
  const float* b3  = (const float*)d_in[7];
  const float* fW1 = (const float*)d_in[8];
  const float* fb1 = (const float*)d_in[9];
  const float* fW2 = (const float*)d_in[10];
  const float* fb2 = (const float*)d_in[11];
  const float* fW3 = (const float*)d_in[12];
  const float* fb3 = (const float*)d_in[13];
  const int*   nbr = (const int*)d_in[14];
  float* out = (float*)d_out;

  dim3 grid(BN / WVB);   // 4096 blocks x 256 threads = 1 wave per point
  axisnet_kernel<<<grid, 256, 0, stream>>>(xyz, local_axis, W1, b1, W2, b2,
                                           W3, b3, fW1, fb1, fW2, fb2, fW3,
                                           fb3, nbr, out);
}
